// SparseTransformerBlock_79551384257120
// MI455X (gfx1250) — compile-verified, matrix operand order checked by reading, not run on hardware
//
#include <hip/hip_runtime.h>
#include <hip/hip_bf16.h>

// ---------------------------------------------------------------------------
// SparseTransformerBlock for gfx1250 (MI455X, wave32, WMMA).
// - All GEMMs use v_wmma_f32_16x16x32_f16 (f16 in, f32 accumulate).
// - Weights are pre-converted to f16 once per call; weight-tile staging uses
//   GLOBAL_LOAD_ASYNC_TO_LDS_B128 (ASYNCcnt path) when the builtin exists,
//   with a synchronous 16B-copy fallback.
// - Problem is tiny vs the chip (~36 GFLOP, ~57MB weights fit in L2), so the
//   goal is the right instruction paths: WMMA + async LDS + split waits.
// ---------------------------------------------------------------------------

typedef __attribute__((ext_vector_type(16))) _Float16 v16h;
typedef __attribute__((ext_vector_type(8)))  _Float16 v8h;
typedef __attribute__((ext_vector_type(8)))  float    v8f;
typedef __attribute__((ext_vector_type(4)))  int      v4i;

#define ATH_AS1 __attribute__((address_space(1)))
#define ATH_AS3 __attribute__((address_space(3)))

#if defined(__AMDGCN__) && __has_builtin(__builtin_amdgcn_global_load_async_to_lds_b128)
#define ATH_ASYNC_LDS 1
#else
#define ATH_ASYNC_LDS 0
#endif

__device__ __forceinline__ void ath_wait_async0()
{
#if defined(__AMDGCN__)
#if __has_builtin(__builtin_amdgcn_s_wait_asynccnt)
  __builtin_amdgcn_s_wait_asynccnt(0);
#else
  asm volatile("s_wait_asynccnt 0x0" ::: "memory");
#endif
#endif
}

// Epilogue modes
//  0: C = alpha * acc
//  1: r = acc + bias[n] + resid[idx]; C = r; C2 = r      (Wo proj + residual)
//  2: C = gelu_exact(acc) * C                            (w1 fused with w3 prod)
//  3: C += rowscale[m*rsStride] * acc                    (w2 weighted accumulate)
// B-source modes (BSRC)
//  0: f32 [N,K] row-major (K for Q@K^T)
//  1: f32 [K,N] row-major, transposed while staging (V for P@V)
//  2: f16 [N,K] row-major, async byte-copy to LDS (all weight matrices)
template <int EPI, int BSRC>
__global__ __launch_bounds__(128)
void gemm_wmma(const float* __restrict__ A, const void* __restrict__ Bv,
               float* __restrict__ C,
               int M, int N, int K, int lda, int ldb, int ldc,
               long aOuter, long aInner, long bOuter, long bInner,
               long cOuter, long cInner, int zdiv,
               const float* __restrict__ bias, const float* __restrict__ resid,
               float* __restrict__ C2, const float* __restrict__ rowscale,
               int rsStride, float alpha)
{
  __shared__ __align__(16) _Float16 As[64 * 40];
  __shared__ __align__(16) _Float16 Bs[64 * 40];

  const float*    Bf = (const float*)Bv;
  const _Float16* Bh = (const _Float16*)Bv;

  // batched offsets: z = zo*zdiv + zi  (e.g. zo=batch, zi=head)
  const int z  = blockIdx.z;
  const int zo = z / zdiv;
  const int zi = z % zdiv;
  A += (long)zo * aOuter + (long)zi * aInner;
  C += (long)zo * cOuter + (long)zi * cInner;
  if constexpr (BSRC == 2) Bh += (long)zo * bOuter + (long)zi * bInner;
  else                     Bf += (long)zo * bOuter + (long)zi * bInner;

  const int m0   = blockIdx.y * 64;
  const int n0   = blockIdx.x * 64;
  const int tid  = threadIdx.x;
  const int lane = tid & 31;
  const int wave = tid >> 5;
  const int wm   = (wave >> 1) * 32;  // wave row offset within block tile
  const int wn   = (wave & 1) * 32;   // wave col offset

  v8f acc[2][2] = {};

  const int frow = lane & 15;          // row (A) / col (B) within 16-tile
  const int kb   = (lane >> 4) << 3;   // 0 for lanes 0-15, 8 for lanes 16-31

  for (int k0 = 0; k0 < K; k0 += 32) {
    // ---- stage B tile into Bs[n][k] ----
    if constexpr (BSRC == 2) {
      // f16 [N,K]: pure byte copy, 64 rows x 64B = 256 x b128 chunks
#if ATH_ASYNC_LDS
#pragma unroll
      for (int i = 0; i < 2; ++i) {
        int lin = tid + 128 * i;   // 0..255
        int r   = lin >> 2;        // row 0..63
        int c16 = lin & 3;         // 16B chunk within row
        const _Float16* gp = Bh + (long)(n0 + r) * ldb + k0 + c16 * 8;
        _Float16*       lp = &Bs[r * 40 + c16 * 8];
        __builtin_amdgcn_global_load_async_to_lds_b128(
            (ATH_AS1 v4i*)(void*)gp, (ATH_AS3 v4i*)(void*)lp, 0, 0);
      }
#else
#pragma unroll
      for (int i = 0; i < 2; ++i) {
        int lin = tid + 128 * i;
        int r   = lin >> 2;
        int c16 = lin & 3;
        const v8h v = *(const v8h*)(Bh + (long)(n0 + r) * ldb + k0 + c16 * 8);
        *(v8h*)&Bs[r * 40 + c16 * 8] = v;
      }
#endif
    } else if constexpr (BSRC == 0) {
      // f32 [N,K] row-major: convert while staging
#pragma unroll
      for (int i = 0; i < 4; ++i) {
        int lin = tid + 128 * i;   // 512 float4 = 64x32 f32
        int r   = lin >> 3;        // n-local 0..63
        int c4  = lin & 7;
        const float4 v = *(const float4*)(Bf + (long)(n0 + r) * ldb + k0 + c4 * 4);
        _Float16* d = &Bs[r * 40 + c4 * 4];
        d[0] = (_Float16)v.x; d[1] = (_Float16)v.y;
        d[2] = (_Float16)v.z; d[3] = (_Float16)v.w;
      }
    } else {
      // f32 [K,N] row-major (V matrix in P@V): transpose while staging
#pragma unroll
      for (int i = 0; i < 4; ++i) {
        int lin = tid + 128 * i;  // 512 float4 = 32x64 f32
        int kr  = lin >> 4;       // k-local 0..31
        int c4  = lin & 15;       // n/4
        const float4 v = *(const float4*)(Bf + (long)(k0 + kr) * ldb + n0 + c4 * 4);
        Bs[(c4 * 4 + 0) * 40 + kr] = (_Float16)v.x;
        Bs[(c4 * 4 + 1) * 40 + kr] = (_Float16)v.y;
        Bs[(c4 * 4 + 2) * 40 + kr] = (_Float16)v.z;
        Bs[(c4 * 4 + 3) * 40 + kr] = (_Float16)v.w;
      }
    }

    // ---- stage A tile (64 x 32), f32 -> f16 into LDS ----
#pragma unroll
    for (int i = 0; i < 4; ++i) {
      int lin = tid + 128 * i;   // 512 float4 = 64x32 f32
      int r   = lin >> 3;        // row 0..63
      int c4  = lin & 7;         // float4 index within row
      const float* gp = A + (long)(m0 + r) * lda + k0 + c4 * 4;
      const float4 v = *(const float4*)gp;
      _Float16* d = &As[r * 40 + c4 * 4];
      d[0] = (_Float16)v.x; d[1] = (_Float16)v.y;
      d[2] = (_Float16)v.z; d[3] = (_Float16)v.w;
      if (i == 0 && k0 + 32 < K)  // prefetch next K-tile of A
        __builtin_prefetch(gp + 32, 0, 0);
    }

#if ATH_ASYNC_LDS
    if constexpr (BSRC == 2) ath_wait_async0();
#endif
    __syncthreads();

    // ---- fragments per ISA layout: halves 0..7 = K kb..kb+7,
    //      halves 8..15 = K kb+16..kb+23 ----
    v16h af[2], bf[2];
#pragma unroll
    for (int i = 0; i < 2; ++i) {
      const _Float16* pa = &As[(wm + i * 16 + frow) * 40 + kb];
      ((v8h*)&af[i])[0] = *(const v8h*)(pa);
      ((v8h*)&af[i])[1] = *(const v8h*)(pa + 16);
      const _Float16* pb = &Bs[(wn + i * 16 + frow) * 40 + kb];
      ((v8h*)&bf[i])[0] = *(const v8h*)(pb);
      ((v8h*)&bf[i])[1] = *(const v8h*)(pb + 16);
    }
#pragma unroll
    for (int i = 0; i < 2; ++i)
#pragma unroll
      for (int j = 0; j < 2; ++j)
        acc[i][j] = __builtin_amdgcn_wmma_f32_16x16x32_f16(
            false, af[i], false, bf[j], (short)0, acc[i][j], false, false);
    __syncthreads();
  }

  // ---- epilogue; D layout: lane&15 = n, vgpr v -> m = v + ((lane>=16)?8:0) ----
  const int nl = lane & 15;
  const int mb = (lane >> 4) << 3;
#pragma unroll
  for (int i = 0; i < 2; ++i)
#pragma unroll
    for (int j = 0; j < 2; ++j)
#pragma unroll
      for (int v = 0; v < 8; ++v) {
        const int  m   = m0 + wm + i * 16 + mb + v;
        const int  n   = n0 + wn + j * 16 + nl;
        const long idx = (long)m * ldc + n;
        const float val = acc[i][j][v];
        if constexpr (EPI == 0) {
          C[idx] = alpha * val;
        } else if constexpr (EPI == 1) {
          const float r = val + bias[n] + resid[idx];
          C[idx]  = r;
          C2[idx] = r;
        } else if constexpr (EPI == 2) {
          const float g = 0.5f * val * (1.0f + erff(val * 0.70710678118654752f));
          C[idx] = g * C[idx];
        } else {
          C[idx] += rowscale[(long)m * rsStride] * val;
        }
      }
}

// ---------------- f32 -> f16 weight conversion ------------------------------
__global__ __launch_bounds__(256)
void f32_to_f16(const float* __restrict__ src, _Float16* __restrict__ dst, long n)
{
  const long base = (long)blockIdx.x * 2048 + threadIdx.x;
#pragma unroll
  for (int j = 0; j < 8; ++j) {
    const long i = base + (long)j * 256;
    if (i < n) dst[i] = (_Float16)src[i];
  }
}

// ---------------- LayerNorm: one block per token (D = 768 = 256*3) ----------
__global__ __launch_bounds__(256)
void layernorm_k(const float* __restrict__ X, const float* __restrict__ g,
                 const float* __restrict__ b, float* __restrict__ Y)
{
  __shared__ float s1[256];
  __shared__ float s2[256];
  const float* x = X + (long)blockIdx.x * 768;
  float*       y = Y + (long)blockIdx.x * 768;
  const int t = threadIdx.x;
  const float v0 = x[t], v1 = x[t + 256], v2 = x[t + 512];
  s1[t] = v0 + v1 + v2;
  s2[t] = v0 * v0 + v1 * v1 + v2 * v2;
  __syncthreads();
  for (int s = 128; s > 0; s >>= 1) {
    if (t < s) { s1[t] += s1[t + s]; s2[t] += s2[t + s]; }
    __syncthreads();
  }
  const float mean = s1[0] * (1.0f / 768.0f);
  const float var  = s2[0] * (1.0f / 768.0f) - mean * mean;
  const float inv  = rsqrtf(var + 1e-5f);
  y[t]       = (v0 - mean) * inv * g[t]       + b[t];
  y[t + 256] = (v1 - mean) * inv * g[t + 256] + b[t + 256];
  y[t + 512] = (v2 - mean) * inv * g[t + 512] + b[t + 512];
}

// ---------------- Row softmax over 512-wide score rows ----------------------
__global__ __launch_bounds__(256)
void softmax_rows(float* __restrict__ S)
{
  __shared__ float red[256];
  float* row = S + (long)blockIdx.x * 512;
  const int t = threadIdx.x;
  const float a = row[t], b = row[t + 256];
  red[t] = fmaxf(a, b);
  __syncthreads();
  for (int s = 128; s > 0; s >>= 1) {
    if (t < s) red[t] = fmaxf(red[t], red[t + s]);
    __syncthreads();
  }
  const float mx = red[0];
  __syncthreads();
  const float ea = expf(a - mx), eb = expf(b - mx);
  red[t] = ea + eb;
  __syncthreads();
  for (int s = 128; s > 0; s >>= 1) {
    if (t < s) red[t] += red[t + s];
    __syncthreads();
  }
  const float inv = 1.0f / red[0];
  row[t]       = ea * inv;
  row[t + 256] = eb * inv;
}

// ---------------- MoE gate: cw[t,e] = softmax((h+feat[m]) @ Wg^T) -----------
// (K == E == 2, so top-k keeps everything and renorm is identity.)
__global__ __launch_bounds__(256)
void gate_k(const float* __restrict__ Hbuf, const long long* __restrict__ mlen,
            const float* __restrict__ Wg, const float* __restrict__ feat,
            float* __restrict__ cw)
{
  __shared__ float s0[256];
  __shared__ float s1[256];
  const int  t   = threadIdx.x;
  const long tok = blockIdx.x;
  const int  n   = (int)(tok % 512);   // position within sequence
  int mi = 0;
  long long accum = 0;
  for (int i = 0; i < 2; ++i) { accum += mlen[i]; if (n >= accum) mi = i + 1; }
  if (mi > 1) mi = 1;
  const float* h = Hbuf + tok * 768;
  const float* f = feat + (long)mi * 768;
  float d0 = 0.0f, d1 = 0.0f;
  for (int i = t; i < 768; i += 256) {
    const float v = h[i] + f[i];
    d0 += v * Wg[i];
    d1 += v * Wg[768 + i];
  }
  s0[t] = d0; s1[t] = d1;
  __syncthreads();
  for (int s = 128; s > 0; s >>= 1) {
    if (t < s) { s0[t] += s0[t + s]; s1[t] += s1[t + s]; }
    __syncthreads();
  }
  if (t == 0) {
    const float l0 = s0[0], l1 = s1[0];
    const float mx = fmaxf(l0, l1);
    const float e0 = expf(l0 - mx), e1 = expf(l1 - mx);
    const float inv = 1.0f / (e0 + e1);
    cw[tok * 2]     = e0 * inv;
    cw[tok * 2 + 1] = e1 * inv;
  }
}

// ---------------------------------------------------------------------------
extern "C" void kernel_launch(void* const* d_in, const int* in_sizes, int n_in,
                              void* d_out, int out_size, void* d_ws, size_t ws_size,
                              hipStream_t stream)
{
  (void)in_sizes; (void)n_in; (void)out_size; (void)ws_size;

  const float*     x    = (const float*)d_in[0];
  const long long* mlen = (const long long*)d_in[1];
  const float*     ln1g = (const float*)d_in[2];
  const float*     ln1b = (const float*)d_in[3];
  const float*     Wq   = (const float*)d_in[4];
  const float*     Wkv  = (const float*)d_in[5];
  const float*     Wo   = (const float*)d_in[6];
  const float*     bo   = (const float*)d_in[7];
  const float*     ln2g = (const float*)d_in[8];
  const float*     ln2b = (const float*)d_in[9];
  const float*     Wg   = (const float*)d_in[10];
  const float*     feat = (const float*)d_in[11];
  const float*     w1   = (const float*)d_in[12];
  const float*     w3   = (const float*)d_in[13];
  const float*     w2   = (const float*)d_in[14];
  float*           out  = (float*)d_out;

  // Workspace layout (floats), liveness-aliased:
  //  region0: xln (LN1 out)        -> reused as ob (attn out)
  //  region1: qb                   -> reused as x1 (post-attn residual)
  //  region2: kvb (1.57M)          -> reused as h2 (LN2 out)
  //  region3: Sb  (6.29M scores)   -> reused as pb (MoE hidden, 3.15M)
  //  region4: cw  (gate weights)
  //  region5: f16 weight copies (16.5M halfs)
  float* w   = (float*)d_ws;
  float* xln = w;                        float* ob = xln;
  float* qb  = w + 786432;               float* x1 = qb;
  float* kvb = w + 2 * 786432;           float* h2 = kvb;
  float* Sb  = w + 2 * 786432 + 1572864; float* pb = Sb;
  float* cw  = Sb + 6291456;

  _Float16* hWq  = (_Float16*)(w + 9439232);
  _Float16* hWkv = hWq  + 589824;
  _Float16* hWo  = hWkv + 1179648;
  _Float16* hw1  = hWo  + 589824;
  _Float16* hw3  = hw1  + 4718592;
  _Float16* hw2  = hw3  + 4718592;

  // 0) convert all weights to f16 (byte-copyable by async LDS staging)
  f32_to_f16<<<288,  256, 0, stream>>>(Wq,  hWq,  589824);
  f32_to_f16<<<576,  256, 0, stream>>>(Wkv, hWkv, 1179648);
  f32_to_f16<<<288,  256, 0, stream>>>(Wo,  hWo,  589824);
  f32_to_f16<<<2304, 256, 0, stream>>>(w1,  hw1,  4718592);
  f32_to_f16<<<2304, 256, 0, stream>>>(w3,  hw3,  4718592);
  f32_to_f16<<<2304, 256, 0, stream>>>(w2,  hw2,  4718592);

  // 1) LN1
  layernorm_k<<<1024, 256, 0, stream>>>(x, ln1g, ln1b, xln);

  // 2) Q = xln @ Wq^T   (M=1024, N=768, K=768)
  gemm_wmma<0, 2><<<dim3(12, 16, 1), 128, 0, stream>>>(
      xln, hWq, qb, 1024, 768, 768, 768, 768, 768,
      0, 0, 0, 0, 0, 0, 1, nullptr, nullptr, nullptr, nullptr, 0, 1.0f);

  // 3) KV = xln @ Wkv^T (M=1024, N=1536, K=768)
  gemm_wmma<0, 2><<<dim3(24, 16, 1), 128, 0, stream>>>(
      xln, hWkv, kvb, 1024, 1536, 768, 768, 768, 1536,
      0, 0, 0, 0, 0, 0, 1, nullptr, nullptr, nullptr, nullptr, 0, 1.0f);

  // 4) S = scale * Q @ K^T, batched over z = b*12 + h (M=N=512, K=64)
  gemm_wmma<0, 0><<<dim3(8, 8, 24), 128, 0, stream>>>(
      qb, kvb, Sb, 512, 512, 64, 768, 1536, 512,
      512L * 768, 64,            // A: +b*N*D, +h*64
      512L * 1536, 64,           // B: keys at kv[:, h*64 + k]
      12L * 512 * 512, 512L * 512,
      12, nullptr, nullptr, nullptr, nullptr, 0, 0.125f);

  // 5) softmax over each of the 24*512 score rows
  softmax_rows<<<12288, 256, 0, stream>>>(Sb);

  // 6) O = P @ V (V is [K,N] row-major), batched (M=512, N=64, K=512)
  gemm_wmma<0, 1><<<dim3(1, 8, 24), 128, 0, stream>>>(
      Sb, kvb + 768, ob, 512, 64, 512, 512, 1536, 768,
      12L * 512 * 512, 512L * 512,
      512L * 1536, 64,
      512L * 768, 64,
      12, nullptr, nullptr, nullptr, nullptr, 0, 1.0f);

  // 7) x1 = O @ Wo^T + bo + x  (also written to d_out)
  gemm_wmma<1, 2><<<dim3(12, 16, 1), 128, 0, stream>>>(
      ob, hWo, x1, 1024, 768, 768, 768, 768, 768,
      0, 0, 0, 0, 0, 0, 1, bo, x, out, nullptr, 0, 1.0f);

  // 8) LN2 -> h2
  layernorm_k<<<1024, 256, 0, stream>>>(x1, ln2g, ln2b, h2);

  // 9) gating weights
  gate_k<<<1024, 256, 0, stream>>>(h2, mlen, Wg, feat, cw);

  // 10) MoE experts (sequential on the stream; pb reused per expert)
  for (int e = 0; e < 2; ++e) {
    const _Float16* w3e = hw3 + (long)e * 3072 * 768;
    const _Float16* w1e = hw1 + (long)e * 3072 * 768;
    const _Float16* w2e = hw2 + (long)e * 768 * 3072;
    // u = h2 @ w3[e]^T
    gemm_wmma<0, 2><<<dim3(48, 16, 1), 128, 0, stream>>>(
        h2, w3e, pb, 1024, 3072, 768, 768, 768, 3072,
        0, 0, 0, 0, 0, 0, 1, nullptr, nullptr, nullptr, nullptr, 0, 1.0f);
    // p = gelu(h2 @ w1[e]^T) * u
    gemm_wmma<2, 2><<<dim3(48, 16, 1), 128, 0, stream>>>(
        h2, w1e, pb, 1024, 3072, 768, 768, 768, 3072,
        0, 0, 0, 0, 0, 0, 1, nullptr, nullptr, nullptr, nullptr, 0, 1.0f);
    // out += cw[:,e] * (p @ w2[e]^T)
    gemm_wmma<3, 2><<<dim3(12, 16, 1), 128, 0, stream>>>(
        pb, w2e, out, 1024, 768, 3072, 3072, 3072, 768,
        0, 0, 0, 0, 0, 0, 1, nullptr, nullptr, nullptr, cw + e, 2, 1.0f);
  }
}
